// WLGNN_15307263442981
// MI455X (gfx1250) — matrix operand order repaired and checked by reading, not compiled, not run on hardware
//
#include <hip/hip_runtime.h>
#include <hip/hip_bf16.h>

typedef __attribute__((ext_vector_type(16))) _Float16 v16h;
typedef __attribute__((ext_vector_type(8)))  _Float16 v8h;
typedef __attribute__((ext_vector_type(8)))  float    v8f;
typedef _Float16 half_t;

#define EPSN 1e-5f

__device__ __forceinline__ v8f wmma_f16(v16h a, v16h b, v8f c) {
    return __builtin_amdgcn_wmma_f32_16x16x32_f16(false, a, false, b, (short)0, c, false, false);
}

// ---------------- mask / z preparation ----------------
__global__ void k_maskset(const int* __restrict__ pos2, int* __restrict__ mask, int P) {
    int j = blockIdx.x * blockDim.x + threadIdx.x;
    if (j < P) mask[pos2[j]] = 1;
}

__global__ void k_count(const int* __restrict__ mask, int* __restrict__ cnt, int n) {
    __shared__ int s;
    if (threadIdx.x == 0) s = 0;
    __syncthreads();
    int l = 0;
    for (int i = blockIdx.x * blockDim.x + threadIdx.x; i < n; i += gridDim.x * blockDim.x)
        l += mask[i];
    atomicAdd(&s, l);
    __syncthreads();
    if (threadIdx.x == 0) atomicAdd(cnt, s);
}

// z0/z1 = the two possible normalized rows of emb2_w under GraphNorm with masked counts
__global__ void k_zprep(const int* __restrict__ cnt, const float* __restrict__ emb2,
                        const float* __restrict__ w, const float* __restrict__ b,
                        const float* __restrict__ ms, float* __restrict__ z0,
                        float* __restrict__ z1, float fM) {
    int f = threadIdx.x;
    if (f >= 16) return;
    float c  = (float)(*cnt);
    float e0 = emb2[f], e1 = emb2[16 + f];
    float mean = (c * e1 + (fM - c) * e0) / fM;
    float ex2  = (c * e1 * e1 + (fM - c) * e0 * e0) / fM;
    float m = ms[f];
    float var = ex2 - mean * mean * m * (2.0f - m);
    float a = w[f] * rsqrtf(var + EPSN);
    float bb = b[f] - a * m * mean;
    z0[f] = a * e0 + bb;
    z1[f] = a * e1 + bb;
}

// ---------------- degree / dinv ----------------
__global__ void k_deg(const int* __restrict__ col, float* __restrict__ deg, int E) {
    for (int e = blockIdx.x * blockDim.x + threadIdx.x; e < E; e += gridDim.x * blockDim.x)
        atomicAdd(&deg[col[e]], 1.0f);
}

__global__ void k_dinv(float* __restrict__ deg, int n) {
    int i = blockIdx.x * blockDim.x + threadIdx.x;
    if (i < n) deg[i] = rsqrtf(deg[i] + 1.0f);
}

// ---------------- embedding gather (store f16 for WMMA A) + per-feature f32 stats ----------------
__global__ void k_emb_gather(const int* __restrict__ x, const float* __restrict__ emb,
                             half_t* __restrict__ h0, float* __restrict__ stats, int n) {
    __shared__ float ssum[32], ssq[32];
    int f = threadIdx.x & 31;
    if (threadIdx.x < 32) { ssum[threadIdx.x] = 0.f; ssq[threadIdx.x] = 0.f; }
    __syncthreads();
    float lsum = 0.f, lsq = 0.f;
    int rpb = blockDim.x >> 5;
    for (int i = blockIdx.x * rpb + (int)(threadIdx.x >> 5); i < n; i += gridDim.x * rpb) {
        int idx = x[i];
        float v = emb[(size_t)idx * 32 + f];
        h0[(size_t)i * 32 + f] = (half_t)v;
        lsum += v; lsq += v * v;
    }
    atomicAdd(&ssum[f], lsum);
    atomicAdd(&ssq[f], lsq);
    __syncthreads();
    if (threadIdx.x < 32) {
        atomicAdd(&stats[threadIdx.x], ssum[threadIdx.x]);
        atomicAdd(&stats[32 + threadIdx.x], ssq[threadIdx.x]);
    }
}

// gn0 stats -> alpha/beta; fold alpha into W, packed in WMMA B-operand lane layout, + betaW row
// WnB[tile(2)][lane(32)][e(16)] : element (k = 8*(lane>>4)+(e&7)+16*(e>>3) , n = tile*16 + (lane&15))
__global__ void k_prep_conv1(const float* __restrict__ stats, const float* __restrict__ gw,
                             const float* __restrict__ gb, const float* __restrict__ gms,
                             const float* __restrict__ W, half_t* __restrict__ WnB,
                             float* __restrict__ betaW, float invN) {
    __shared__ float sa[32], sb[32];
    int t = threadIdx.x;                // 1024 threads
    if (t < 32) {
        float mean = stats[t] * invN;
        float ex2  = stats[32 + t] * invN;
        float m = gms[t];
        float var = ex2 - mean * mean * m * (2.0f - m);
        float a = gw[t] * rsqrtf(var + EPSN);
        sa[t] = a;
        sb[t] = gb[t] - a * m * mean;
    }
    __syncthreads();
    int tile = t >> 9, L = (t >> 4) & 31, e = t & 15;
    int k  = 8 * (L >> 4) + (e & 7) + 16 * (e >> 3);
    int nn = tile * 16 + (L & 15);
    WnB[t] = (half_t)(sa[k] * W[k * 32 + nn]);
    if (t < 32) {
        float acc = 0.f;
        for (int kk = 0; kk < 32; ++kk) acc += sb[kk] * W[kk * 32 + t];
        betaW[t] = acc;
    }
}

// pack c2_W [48x32] into B lane layout, K padded to 64:
// W2B[c(4)][lane(32)][e(16)], c = chunk*2 + tile
__global__ void k_prep_conv2(const float* __restrict__ W, half_t* __restrict__ W2B) {
    int t = blockIdx.x * blockDim.x + threadIdx.x;   // 2048 threads
    if (t >= 2048) return;
    int c = t >> 9;
    int chunk = c >> 1, tile = c & 1;
    int L = (t >> 4) & 31, e = t & 15;
    int kl = 8 * (L >> 4) + (e & 7) + 16 * (e >> 3);
    int nn = tile * 16 + (L & 15);
    float v;
    if (chunk == 0) v = W[kl * 32 + nn];
    else            v = (kl < 16) ? W[(32 + kl) * 32 + nn] : 0.f;
    W2B[t] = (half_t)v;
}

// ---------------- WMMA GEMM 1: hW1 = h0f16 @ WnB + betaW   ([N,32]x[32,32]) ----------------
__global__ void k_gemm1(const half_t* __restrict__ h0, const half_t* __restrict__ WnB,
                        const float* __restrict__ betaW, float* __restrict__ hW, int n) {
    int lane = threadIdx.x & 31;
    int wv = (blockIdx.x * blockDim.x + threadIdx.x) >> 5;
    int base = wv * 16;
    if (base >= n) return;                 // n is a multiple of 16: full tiles only
    int m = lane & 15, hi = lane >> 4;
    int row = base + m; if (row >= n) row = n - 1;
    const half_t* rp = h0 + (size_t)row * 32;
    v8h alo = *(const v8h*)(rp + 8 * hi);
    v8h ahi = *(const v8h*)(rp + 16 + 8 * hi);
    v16h a;
    #pragma unroll
    for (int e = 0; e < 8; ++e) { a[e] = alo[e]; a[8 + e] = ahi[e]; }
    const half_t* bp = WnB + (size_t)lane * 16;
    v8h b0a = *(const v8h*)(bp);       v8h b0b = *(const v8h*)(bp + 8);
    v8h b1a = *(const v8h*)(bp + 512); v8h b1b = *(const v8h*)(bp + 520);
    v16h b0, b1;
    #pragma unroll
    for (int e = 0; e < 8; ++e) {
        b0[e] = b0a[e]; b0[8 + e] = b0b[e];
        b1[e] = b1a[e]; b1[8 + e] = b1b[e];
    }
    v8f c0 = {}, c1 = {};
    c0 = wmma_f16(a, b0, c0);
    c1 = wmma_f16(a, b1, c1);
    float bw0 = betaW[m], bw1 = betaW[16 + m];
    float* op = hW + (size_t)(base + 8 * hi) * 32 + m;
    #pragma unroll
    for (int r = 0; r < 8; ++r) {
        op[(size_t)r * 32]      = c0[r] + bw0;
        op[(size_t)r * 32 + 16] = c1[r] + bw1;
    }
}

// ---------------- edge scatter-add: agg[col] += hW[row]*dinv[row]*dinv[col] ----------------
__global__ void k_edge_agg(const int* __restrict__ row, const int* __restrict__ col,
                           const float* __restrict__ hW, const float* __restrict__ dinv,
                           float* __restrict__ agg, int E) {
    size_t total = (size_t)E * 32;
    for (size_t t = (size_t)blockIdx.x * blockDim.x + threadIdx.x; t < total;
         t += (size_t)gridDim.x * blockDim.x) {
        int e = (int)(t >> 5), f = (int)(t & 31);
        int r = row[e], c = col[e];
        float coef = dinv[r] * dinv[c];
        atomicAdd(&agg[(size_t)c * 32 + f], hW[(size_t)r * 32 + f] * coef);
    }
}

// y = agg + hW*dinv^2 + bias (in place), float4-vectorized, + per-feature stats
__global__ void k_selfloop_stats(float* __restrict__ agg, const float* __restrict__ hW,
                                 const float* __restrict__ dinv, const float* __restrict__ bias,
                                 float* __restrict__ stats, int n) {
    __shared__ float ssum[32], ssq[32];
    if (threadIdx.x < 32) { ssum[threadIdx.x] = 0.f; ssq[threadIdx.x] = 0.f; }
    __syncthreads();
    int fb = (threadIdx.x & 7) * 4;
    float4 b4 = *(const float4*)(bias + fb);
    float ls0 = 0.f, ls1 = 0.f, ls2 = 0.f, ls3 = 0.f;
    float lq0 = 0.f, lq1 = 0.f, lq2 = 0.f, lq3 = 0.f;
    size_t total = (size_t)n * 8;
    for (size_t t = (size_t)blockIdx.x * blockDim.x + threadIdx.x; t < total;
         t += (size_t)gridDim.x * blockDim.x) {
        int i = (int)(t >> 3);
        size_t off = (size_t)i * 32 + ((t & 7) << 2);
        float d = dinv[i]; float d2 = d * d;
        float4 a4 = *(float4*)(agg + off);
        float4 h4 = *(const float4*)(hW + off);
        float v0 = a4.x + h4.x * d2 + b4.x;
        float v1 = a4.y + h4.y * d2 + b4.y;
        float v2 = a4.z + h4.z * d2 + b4.z;
        float v3 = a4.w + h4.w * d2 + b4.w;
        *(float4*)(agg + off) = make_float4(v0, v1, v2, v3);
        ls0 += v0; lq0 += v0 * v0;
        ls1 += v1; lq1 += v1 * v1;
        ls2 += v2; lq2 += v2 * v2;
        ls3 += v3; lq3 += v3 * v3;
    }
    atomicAdd(&ssum[fb + 0], ls0); atomicAdd(&ssq[fb + 0], lq0);
    atomicAdd(&ssum[fb + 1], ls1); atomicAdd(&ssq[fb + 1], lq1);
    atomicAdd(&ssum[fb + 2], ls2); atomicAdd(&ssq[fb + 2], lq2);
    atomicAdd(&ssum[fb + 3], ls3); atomicAdd(&ssq[fb + 3], lq3);
    __syncthreads();
    if (threadIdx.x < 32) {
        atomicAdd(&stats[threadIdx.x], ssum[threadIdx.x]);
        atomicAdd(&stats[32 + threadIdx.x], ssq[threadIdx.x]);
    }
}

__global__ void k_norm_coef(const float* __restrict__ ssum, const float* __restrict__ ssq,
                            const float* __restrict__ w, const float* __restrict__ b,
                            const float* __restrict__ ms, float* __restrict__ alpha,
                            float* __restrict__ beta, float invN) {
    int f = threadIdx.x;
    float mean = ssum[f] * invN;
    float ex2  = ssq[f] * invN;
    float m = ms[f];
    float var = ex2 - mean * mean * m * (2.0f - m);
    float a = w[f] * rsqrtf(var + EPSN);
    alpha[f] = a;
    beta[f]  = b[f] - a * m * mean;
}

// cat[j,0:32] = relu(a*y1[p0]+b)+relu(a*y1[p1]+b); cat[j,32:48] = z0/z1; stored f16 for GEMM2
__global__ void k_pairs(const int* __restrict__ pos1, const int* __restrict__ mask,
                        const float* __restrict__ y1, const float* __restrict__ alpha,
                        const float* __restrict__ beta, const float* __restrict__ z0,
                        const float* __restrict__ z1, half_t* __restrict__ cat, int m) {
    size_t total = (size_t)m * 48;
    for (size_t t = (size_t)blockIdx.x * blockDim.x + threadIdx.x; t < total;
         t += (size_t)gridDim.x * blockDim.x) {
        int j = (int)(t / 48);
        int f = (int)(t - (size_t)j * 48);
        float v;
        if (f < 32) {
            int a0 = pos1[2 * j], b0 = pos1[2 * j + 1];
            float va = alpha[f] * y1[(size_t)a0 * 32 + f] + beta[f];
            float vb = alpha[f] * y1[(size_t)b0 * 32 + f] + beta[f];
            v = fmaxf(va, 0.f) + fmaxf(vb, 0.f);
        } else {
            int q = f - 32;
            v = mask[j] ? z1[q] : z0[q];
        }
        cat[t] = (half_t)v;
    }
}

// ---------------- WMMA GEMM 2: hW2 = catf16 @ W2B  ([M,48]x[48,32], K padded to 64) ----------------
__global__ void k_gemm2(const half_t* __restrict__ cat, const half_t* __restrict__ W2B,
                        float* __restrict__ hW, int n) {
    int lane = threadIdx.x & 31;
    int wv = (blockIdx.x * blockDim.x + threadIdx.x) >> 5;
    int base = wv * 16;
    if (base >= n) return;                 // n is a multiple of 16
    int m = lane & 15, hi = lane >> 4;
    int row = base + m; if (row >= n) row = n - 1;
    const half_t* rp = cat + (size_t)row * 48;
    v8h a0lo = *(const v8h*)(rp + 8 * hi);
    v8h a0hi = *(const v8h*)(rp + 16 + 8 * hi);
    v8h a1lo = *(const v8h*)(rp + 32 + 8 * hi);
    v16h a0, a1;
    #pragma unroll
    for (int e = 0; e < 8; ++e) {
        a0[e] = a0lo[e]; a0[8 + e] = a0hi[e];
        a1[e] = a1lo[e]; a1[8 + e] = (half_t)0.0f;   // K 48..63 padding
    }
    const half_t* bp = W2B + (size_t)lane * 16;
    v16h b[4];
    #pragma unroll
    for (int c = 0; c < 4; ++c) {
        v8h x0 = *(const v8h*)(bp + (size_t)c * 512);
        v8h x1 = *(const v8h*)(bp + (size_t)c * 512 + 8);
        #pragma unroll
        for (int e = 0; e < 8; ++e) { b[c][e] = x0[e]; b[c][8 + e] = x1[e]; }
    }
    v8f c0 = {}, c1 = {};
    c0 = wmma_f16(a0, b[0], c0);   // chunk0, cols 0-15
    c1 = wmma_f16(a0, b[1], c1);   // chunk0, cols 16-31
    c0 = wmma_f16(a1, b[2], c0);   // chunk1, cols 0-15
    c1 = wmma_f16(a1, b[3], c1);   // chunk1, cols 16-31
    float* op = hW + (size_t)(base + 8 * hi) * 32 + m;
    #pragma unroll
    for (int r = 0; r < 8; ++r) {
        op[(size_t)r * 32]      = c0[r];
        op[(size_t)r * 32 + 16] = c1[r];
    }
}

// out[j] = relu(alpha*y2[pos2[j]]+beta) . pred_W + pred_b  (one wave per output)
__global__ void k_out(const int* __restrict__ pos2, const float* __restrict__ y2,
                      const float* __restrict__ alpha, const float* __restrict__ beta,
                      const float* __restrict__ pW, const float* __restrict__ pb,
                      float* __restrict__ out, int P) {
    int lane = threadIdx.x & 31;
    int j = (blockIdx.x * blockDim.x + threadIdx.x) >> 5;
    if (j >= P) return;
    int i = pos2[j];
    float v = fmaxf(alpha[lane] * y2[(size_t)i * 32 + lane] + beta[lane], 0.f) * pW[lane];
    #pragma unroll
    for (int s = 16; s; s >>= 1) v += __shfl_xor(v, s, 32);
    if (lane == 0) out[j] = v + pb[0];
}

extern "C" void kernel_launch(void* const* d_in, const int* in_sizes, int n_in,
                              void* d_out, int out_size, void* d_ws, size_t ws_size,
                              hipStream_t stream) {
    const int*   x      = (const int*)d_in[0];
    const int*   edge1  = (const int*)d_in[1];
    const int*   edge2  = (const int*)d_in[2];
    const int*   pos1   = (const int*)d_in[3];
    const int*   pos2   = (const int*)d_in[4];
    const float* emb1_w = (const float*)d_in[5];
    const float* gn0_w  = (const float*)d_in[6];
    const float* gn0_b  = (const float*)d_in[7];
    const float* gn0_ms = (const float*)d_in[8];
    const float* c1_W   = (const float*)d_in[9];
    const float* c1_b   = (const float*)d_in[10];
    const float* gn1_w  = (const float*)d_in[11];
    const float* gn1_b  = (const float*)d_in[12];
    const float* gn1_ms = (const float*)d_in[13];
    const float* emb2_w = (const float*)d_in[14];
    const float* gn2_w  = (const float*)d_in[15];
    const float* gn2_b  = (const float*)d_in[16];
    const float* gn2_ms = (const float*)d_in[17];
    const float* c2_W   = (const float*)d_in[18];
    const float* c2_b   = (const float*)d_in[19];
    const float* gn3_w  = (const float*)d_in[20];
    const float* gn3_b  = (const float*)d_in[21];
    const float* gn3_ms = (const float*)d_in[22];
    const float* pred_W = (const float*)d_in[23];
    const float* pred_b = (const float*)d_in[24];

    const int N  = in_sizes[0];
    const int E1 = in_sizes[1] / 2;
    const int E2 = in_sizes[2] / 2;
    const int M  = in_sizes[3] / 2;
    const int P  = in_sizes[4];

    float* ws = (float*)d_ws;
    size_t H0    = 0;                            // half_t[N*32]  -> N*16 floats
    size_t HW1   = H0   + (size_t)N * 16;
    size_t AGG1  = HW1  + (size_t)N * 32;
    size_t DEG1  = AGG1 + (size_t)N * 32;
    size_t CAT   = DEG1 + (size_t)N;             // half_t[M*48]  -> M*24 floats
    size_t HW2   = CAT  + (size_t)M * 24;
    size_t AGG2  = HW2  + (size_t)M * 32;
    size_t DEG2  = AGG2 + (size_t)M * 32;
    size_t MASK  = DEG2 + (size_t)M;
    size_t STATS = MASK + (size_t)M;
    size_t AB    = STATS + 256;
    size_t WNB   = AB + 256;                     // 1024 halves
    size_t W2BO  = WNB + 512;                    // 2048 halves

    half_t* h0   = (half_t*)(ws + H0);
    float* hW1   = ws + HW1;
    float* agg1  = ws + AGG1;   // becomes y1 in place
    float* dinv1 = ws + DEG1;
    half_t* cat  = (half_t*)(ws + CAT);
    float* hW2   = ws + HW2;
    float* agg2  = ws + AGG2;   // becomes y2 in place
    float* dinv2 = ws + DEG2;
    int*   mask  = (int*)(ws + MASK);
    float* stats = ws + STATS;  // [s0 sum|sq 64][s1 64][s3 64][cnt]
    float* ab    = ws + AB;     // [betaW1 32][a1 32][b1 32][a3 32][b3 32][z0 16][z1 16]
    half_t* WnB  = (half_t*)(ws + WNB);
    half_t* W2B  = (half_t*)(ws + W2BO);
    int*   cnt   = (int*)(stats + 192);

    hipMemsetAsync(stats, 0, 256 * sizeof(float), stream);
    hipMemsetAsync(dinv1, 0, (size_t)N * sizeof(float), stream);
    hipMemsetAsync(dinv2, 0, (size_t)M * sizeof(float), stream);
    hipMemsetAsync(mask,  0, (size_t)M * sizeof(int), stream);
    hipMemsetAsync(agg1,  0, (size_t)N * 32 * sizeof(float), stream);
    hipMemsetAsync(agg2,  0, (size_t)M * 32 * sizeof(float), stream);

    // mask / z rows
    k_maskset<<<(P + 255) / 256, 256, 0, stream>>>(pos2, mask, P);
    k_count<<<256, 256, 0, stream>>>(mask, cnt, M);
    k_zprep<<<1, 16, 0, stream>>>(cnt, emb2_w, gn2_w, gn2_b, gn2_ms, ab + 160, ab + 176, (float)M);

    // degrees -> dinv
    k_deg<<<1024, 256, 0, stream>>>(edge1 + E1, dinv1, E1);
    k_deg<<<1024, 256, 0, stream>>>(edge2 + E2, dinv2, E2);
    k_dinv<<<(N + 255) / 256, 256, 0, stream>>>(dinv1, N);
    k_dinv<<<(M + 255) / 256, 256, 0, stream>>>(dinv2, M);

    // stage 1: embedding gather + gn0 stats; fold norm into packed weights
    k_emb_gather<<<512, 256, 0, stream>>>(x, emb1_w, h0, stats, N);
    k_prep_conv1<<<1, 1024, 0, stream>>>(stats, gn0_w, gn0_b, gn0_ms, c1_W, WnB, ab + 0,
                                         1.0f / (float)N);
    k_prep_conv2<<<2, 1024, 0, stream>>>(c2_W, W2B);

    // conv1
    int waves1 = (N + 15) / 16;
    k_gemm1<<<(waves1 * 32 + 255) / 256, 256, 0, stream>>>(h0, WnB, ab + 0, hW1, N);
    k_edge_agg<<<2048, 256, 0, stream>>>(edge1, edge1 + E1, hW1, dinv1, agg1, E1);
    k_selfloop_stats<<<1024, 256, 0, stream>>>(agg1, hW1, dinv1, c1_b, stats + 64, N);
    k_norm_coef<<<1, 32, 0, stream>>>(stats + 64, stats + 96, gn1_w, gn1_b, gn1_ms,
                                      ab + 32, ab + 64, 1.0f / (float)N);

    // pair-sum pooling + concat with z (f16 for GEMM2)
    k_pairs<<<4096, 256, 0, stream>>>(pos1, mask, agg1, ab + 32, ab + 64, ab + 160, ab + 176,
                                      cat, M);

    // conv2
    int waves2 = (M + 15) / 16;
    k_gemm2<<<(waves2 * 32 + 255) / 256, 256, 0, stream>>>(cat, W2B, hW2, M);
    k_edge_agg<<<4096, 256, 0, stream>>>(edge2, edge2 + E2, hW2, dinv2, agg2, E2);
    k_selfloop_stats<<<1024, 256, 0, stream>>>(agg2, hW2, dinv2, c2_b, stats + 128, M);
    k_norm_coef<<<1, 32, 0, stream>>>(stats + 128, stats + 160, gn3_w, gn3_b, gn3_ms,
                                      ab + 96, ab + 128, 1.0f / (float)M);

    // prediction head
    k_out<<<(P * 32 + 255) / 256, 256, 0, stream>>>(pos2, agg2, ab + 96, ab + 128,
                                                    pred_W, pred_b, (float*)d_out, P);
}
